// GINENet_90048284328640
// MI455X (gfx1250) — matrix operand order repaired and checked by reading, not compiled, not run on hardware
//
#include <hip/hip_runtime.h>
#include <hip/hip_bf16.h>
#include <stdint.h>

#define N_NODES 40000
#define N_EDGES 640000
#define F_INCH  64
#define H_DIM   128
#define G_BATCH 256
#define BN_EPS  1e-5f

typedef __attribute__((ext_vector_type(16))) __bf16        v16bf;
typedef __attribute__((ext_vector_type(8)))  float         v8f;
typedef __attribute__((ext_vector_type(4)))  unsigned int  u32x4;

union BFrag { u32x4 q[2]; v16bf v; };

__device__ __forceinline__ unsigned short f2bf(float x) {
  union { float f; unsigned u; } v; v.f = x;
  unsigned r = v.u + 0x7fffu + ((v.u >> 16) & 1u);   // round-to-nearest-even
  return (unsigned short)(r >> 16);
}

// ---------------------------------------------------------------------------
// Weight packing: fp32 -> bf16, pre-swizzled into WMMA B-fragment order.
// Fragment (k0,n0) covers K rows k0..k0+31, cols n0..n0+15. Per ISA 7.12.2:
// lanes 0-15 hold K=k0..k0+7,k0+16..k0+23, lanes 16-31 hold K=k0+8..k0+15,k0+24..k0+31; VGPR v holds
// the (2v,2v+1) pair. We store lane-major, 16 contiguous bf16 per lane.
// ---------------------------------------------------------------------------
struct PackArgs {
  const float* W1[3]; const float* b1[3];
  const float* gamma[3]; const float* beta[3];
  const float* mean[3]; const float* var[3];
  const float* W2[3];
  const float* lin1W;
  unsigned short* bf; float* b1f;
};

__device__ __forceinline__ void packB(unsigned short* dst, const float* src,
                                      const float* gamma, const float* var,
                                      int K, int N, int t) {
  int f    = t >> 9;          // fragment id (512 elems each)
  int r    = t & 511;
  int lane = r >> 4;
  int s    = r & 15;
  int nt   = N >> 4;
  int k0   = (f / nt) << 5;
  int n0   = (f % nt) << 4;
  int hi   = lane >> 4;
  int k    = k0 + (s < 8 ? hi * 8 + s : 16 + hi * 8 + (s - 8));
  int n    = n0 + (lane & 15);
  float v  = src[k * N + n];
  if (gamma) v *= gamma[n] * rsqrtf(var[n] + BN_EPS);  // fold BatchNorm scale
  dst[t] = f2bf(v);
}

__global__ __launch_bounds__(256) void pack_weights_kernel(PackArgs a) {
  int t = blockIdx.x * 256 + threadIdx.x;
  if (t < 8192)  { packB(a.bf +     0, a.W1[0], a.gamma[0], a.var[0],  64, 128, t); return; }
  t -= 8192;
  if (t < 16384) { packB(a.bf +  8192, a.W1[1], a.gamma[1], a.var[1], 128, 128, t); return; }
  t -= 16384;
  if (t < 16384) { packB(a.bf + 24576, a.W1[2], a.gamma[2], a.var[2], 128, 128, t); return; }
  t -= 16384;
  if (t < 16384) { packB(a.bf + 40960, a.W2[0], nullptr, nullptr, 128, 128, t); return; }
  t -= 16384;
  if (t < 16384) { packB(a.bf + 57344, a.W2[1], nullptr, nullptr, 128, 128, t); return; }
  t -= 16384;
  if (t < 16384) { packB(a.bf + 73728, a.W2[2], nullptr, nullptr, 128, 128, t); return; }
  t -= 16384;
  if (t < 147456){ packB(a.bf + 90112, a.lin1W, nullptr, nullptr, 384, 384, t); return; }
  t -= 147456;
  if (t < 384) {                       // folded BN bias: 3 layers x 128
    int l = t / 128, n = t - l * 128;
    float s = a.gamma[l][n] * rsqrtf(a.var[l][n] + BN_EPS);
    a.b1f[t] = (a.b1[l][n] - a.mean[l][n]) * s + a.beta[l][n];
  }
}

__global__ void zero_kernel(float* __restrict__ p, size_t n) {
  size_t i = (size_t)blockIdx.x * blockDim.x + threadIdx.x;
  if (i < n) p[i] = 0.f;
}

// ---------------------------------------------------------------------------
// Edge messages: msg = relu(x[src] + edge_attr @ We + be), atomic scatter-add.
// C/4 threads per edge, each handles 4 channels; We staged in LDS.
// ---------------------------------------------------------------------------
__global__ __launch_bounds__(256) void edge_msg_kernel(
    const float* __restrict__ x, const int* __restrict__ ei,
    const float* __restrict__ ea, const float* __restrict__ We,
    const float* __restrict__ be, float* __restrict__ aggr,
    int C, int lg) {
  __shared__ float WeS[16 * 128];
  for (int i = threadIdx.x; i < 16 * C; i += 256) WeS[i] = We[i];
  __syncthreads();

  long gt = (long)blockIdx.x * 256 + threadIdx.x;
  int  e  = (int)(gt >> lg);
  if (e >= N_EDGES) return;
  int ch = ((int)gt & ((1 << lg) - 1)) << 2;

  int s = ei[e], d = ei[N_EDGES + e];
  const float* eap = ea + (size_t)e * 16;
  float m0 = be[ch], m1 = be[ch + 1], m2 = be[ch + 2], m3 = be[ch + 3];
#pragma unroll
  for (int k = 0; k < 16; ++k) {
    float a = eap[k];
    const float* wr = &WeS[k * C + ch];
    m0 += a * wr[0]; m1 += a * wr[1]; m2 += a * wr[2]; m3 += a * wr[3];
  }
  const float* xs = x + (size_t)s * C + ch;
  m0 += xs[0]; m1 += xs[1]; m2 += xs[2]; m3 += xs[3];
  m0 = m0 > 0.f ? m0 : 0.f; m1 = m1 > 0.f ? m1 : 0.f;
  m2 = m2 > 0.f ? m2 : 0.f; m3 = m3 > 0.f ? m3 : 0.f;

  float* ap = aggr + (size_t)d * C + ch;
  __hip_atomic_fetch_add(ap + 0, m0, __ATOMIC_RELAXED, __HIP_MEMORY_SCOPE_AGENT);
  __hip_atomic_fetch_add(ap + 1, m1, __ATOMIC_RELAXED, __HIP_MEMORY_SCOPE_AGENT);
  __hip_atomic_fetch_add(ap + 2, m2, __ATOMIC_RELAXED, __HIP_MEMORY_SCOPE_AGENT);
  __hip_atomic_fetch_add(ap + 3, m3, __ATOMIC_RELAXED, __HIP_MEMORY_SCOPE_AGENT);
}

// ---------------------------------------------------------------------------
// WMMA fragment loaders. A: bf16 row-major in LDS (ISA 7.12.2 16-bit A 16x32).
// B: pre-swizzled, 16 contiguous bf16 per lane.
// ---------------------------------------------------------------------------
__device__ __forceinline__ v16bf load_a_lds(const unsigned short* base, int ld,
                                            int mrow, int k0, int lane) {
  BFrag a;
  const unsigned short* ap = base + (size_t)(mrow + (lane & 15)) * ld
                                  + k0 + (lane >> 4) * 8;
  a.q[0] = *(const u32x4*)ap;          // K = k0+hi*8    .. +7
  a.q[1] = *(const u32x4*)(ap + 16);   // K = k0+16+hi*8 .. +7
  return a.v;
}

__device__ __forceinline__ v16bf load_b_packed(const unsigned short* Bp,
                                               int ktile, int ntile,
                                               int ntiles_total, int lane) {
  BFrag b;
  const unsigned short* bp = Bp + (((size_t)ktile * ntiles_total + ntile) << 9)
                                + lane * 16;
  b.q[0] = *(const u32x4*)bp;
  b.q[1] = *(const u32x4*)(bp + 8);
  return b.v;
}

// ---------------------------------------------------------------------------
// Fused node MLP: out = relu( relu(bn_folded((x+aggr)@W1+b1)) @ W2 + b2 )
// 256 threads = 8 waves; tile 64 rows x 128 cols; bf16 WMMA, f32 accum.
// ---------------------------------------------------------------------------
__global__ __launch_bounds__(256) void node_mlp_kernel(
    const float* __restrict__ xin, const float* __restrict__ aggr,
    const unsigned short* __restrict__ W1p, const float* __restrict__ b1f,
    const unsigned short* __restrict__ W2p, const float* __restrict__ b2,
    float* __restrict__ out, int Cin) {
  __shared__ unsigned short Abuf[64 * 128];
  __shared__ unsigned short Tbuf[64 * 128];
  const int tid  = threadIdx.x;
  const int row0 = blockIdx.x * 64;

  if (tid < 64) {                                  // warm weight tables
    __builtin_prefetch(W1p + tid * 256, 0, 0);
    __builtin_prefetch(W2p + tid * 256, 0, 0);
  }
  for (int i = tid; i < 64 * Cin; i += 256) {      // stage A = x + aggr (bf16)
    int r = i / Cin, k = i - r * Cin;
    size_t g = (size_t)(row0 + r) * Cin + k;
    Abuf[i] = f2bf(xin[g] + aggr[g]);
  }
  __syncthreads();

  const int w = tid >> 5, lane = tid & 31;
  const int rt = w >> 1, cg = w & 1;               // 4 row tiles x 2 col groups
  const int nl = lane & 15, hi = lane >> 4;

  // ---- GEMM1: (64 x Cin) @ (Cin x 128), BN folded into W1p/b1f ----
  v8f acc[4] = {};
  for (int k0 = 0; k0 < Cin; k0 += 32) {
    v16bf a = load_a_lds(Abuf, Cin, rt * 16, k0, lane);
#pragma unroll
    for (int ct = 0; ct < 4; ++ct) {
      v16bf b = load_b_packed(W1p, k0 >> 5, cg * 4 + ct, 8, lane);
      acc[ct] = __builtin_amdgcn_wmma_f32_16x16x32_bf16(
          false, a, false, b, (short)0, acc[ct], false, false);
    }
  }
#pragma unroll
  for (int ct = 0; ct < 4; ++ct) {
    int col = (cg * 4 + ct) * 16 + nl;
    float bb = b1f[col];
#pragma unroll
    for (int i = 0; i < 8; ++i) {                  // C layout: VGPR i -> M=i+8*hi
      float v = acc[ct][i] + bb;
      v = v > 0.f ? v : 0.f;
      Tbuf[(rt * 16 + hi * 8 + i) * 128 + col] = f2bf(v);
    }
  }
  __syncthreads();

  // ---- GEMM2: (64 x 128) @ (128 x 128) ----
  v8f dcc[4] = {};
#pragma unroll
  for (int k0 = 0; k0 < 128; k0 += 32) {
    v16bf a = load_a_lds(Tbuf, 128, rt * 16, k0, lane);
#pragma unroll
    for (int ct = 0; ct < 4; ++ct) {
      v16bf b = load_b_packed(W2p, k0 >> 5, cg * 4 + ct, 8, lane);
      dcc[ct] = __builtin_amdgcn_wmma_f32_16x16x32_bf16(
          false, a, false, b, (short)0, dcc[ct], false, false);
    }
  }
#pragma unroll
  for (int ct = 0; ct < 4; ++ct) {
    int col = (cg * 4 + ct) * 16 + nl;
    float bb = b2[col];
#pragma unroll
    for (int i = 0; i < 8; ++i) {
      float v = dcc[ct][i] + bb;
      v = v > 0.f ? v : 0.f;
      out[(size_t)(row0 + rt * 16 + hi * 8 + i) * 128 + col] = v;
    }
  }
}

// ---------------------------------------------------------------------------
// Graph pooling: p[batch[n], off:off+128] += h[n, :]
// ---------------------------------------------------------------------------
__global__ __launch_bounds__(256) void pool_kernel(
    const float* __restrict__ h, const int* __restrict__ batch,
    float* __restrict__ p, int off) {
  int t = blockIdx.x * 256 + threadIdx.x;
  int n = t >> 5;
  if (n >= N_NODES) return;
  int ch = (t & 31) << 2;
  int g = batch[n];
  const float* hp = h + (size_t)n * 128 + ch;
  float* pp = p + (size_t)g * 384 + off + ch;
  __hip_atomic_fetch_add(pp + 0, hp[0], __ATOMIC_RELAXED, __HIP_MEMORY_SCOPE_AGENT);
  __hip_atomic_fetch_add(pp + 1, hp[1], __ATOMIC_RELAXED, __HIP_MEMORY_SCOPE_AGENT);
  __hip_atomic_fetch_add(pp + 2, hp[2], __ATOMIC_RELAXED, __HIP_MEMORY_SCOPE_AGENT);
  __hip_atomic_fetch_add(pp + 3, hp[3], __ATOMIC_RELAXED, __HIP_MEMORY_SCOPE_AGENT);
}

// ---------------------------------------------------------------------------
// Readout GEMM (lin1): q = relu(p @ lin1W + lin1b); M=256,K=384,N=384.
// ---------------------------------------------------------------------------
__global__ __launch_bounds__(256) void gemm_relu_kernel(
    const float* __restrict__ A, const unsigned short* __restrict__ Bp,
    const float* __restrict__ bias, float* __restrict__ C,
    int K, int Ncols) {
  __shared__ unsigned short Abuf[64 * 384];
  const int tid  = threadIdx.x;
  const int row0 = blockIdx.x * 64;
  const int col0 = blockIdx.y * 128;

  for (int i = tid; i < 64 * K; i += 256) {
    int r = i / K, k = i - r * K;
    Abuf[i] = f2bf(A[(size_t)(row0 + r) * K + k]);
  }
  __syncthreads();

  const int w = tid >> 5, lane = tid & 31;
  const int rt = w >> 1, cg = w & 1;
  const int nl = lane & 15, hi = lane >> 4;
  const int ntt = Ncols >> 4;

  v8f acc[4] = {};
  for (int k0 = 0; k0 < K; k0 += 32) {
    v16bf a = load_a_lds(Abuf, K, rt * 16, k0, lane);
#pragma unroll
    for (int ct = 0; ct < 4; ++ct) {
      v16bf b = load_b_packed(Bp, k0 >> 5, (col0 >> 4) + cg * 4 + ct, ntt, lane);
      acc[ct] = __builtin_amdgcn_wmma_f32_16x16x32_bf16(
          false, a, false, b, (short)0, acc[ct], false, false);
    }
  }
#pragma unroll
  for (int ct = 0; ct < 4; ++ct) {
    int col = col0 + (cg * 4 + ct) * 16 + nl;
    float bb = bias[col];
#pragma unroll
    for (int i = 0; i < 8; ++i) {
      float v = acc[ct][i] + bb;
      v = v > 0.f ? v : 0.f;
      C[(size_t)(row0 + rt * 16 + hi * 8 + i) * Ncols + col] = v;
    }
  }
}

// lin2: out[g] = dot(q[g,:384], W) + b   (one wave32 per graph)
__global__ __launch_bounds__(32) void lin2_kernel(
    const float* __restrict__ q, const float* __restrict__ W,
    const float* __restrict__ b, float* __restrict__ out) {
  int g = blockIdx.x, lane = threadIdx.x;
  float s = 0.f;
#pragma unroll
  for (int k = 0; k < 12; ++k) {
    int idx = lane + k * 32;
    s += q[(size_t)g * 384 + idx] * W[idx];
  }
#pragma unroll
  for (int off = 16; off > 0; off >>= 1) s += __shfl_down(s, off, 32);
  if (lane == 0) out[g] = s + b[0];
}

// ---------------------------------------------------------------------------
extern "C" void kernel_launch(void* const* d_in, const int* in_sizes, int n_in,
                              void* d_out, int out_size, void* d_ws, size_t ws_size,
                              hipStream_t stream) {
  const float* x     = (const float*)d_in[0];
  const int*   ei    = (const int*)d_in[1];
  const float* ea    = (const float*)d_in[2];
  const int*   batch = (const int*)d_in[3];
  auto cp = [&](int l, int j) { return (const float*)d_in[4 + 10 * l + j]; };
  const float* lin1W = (const float*)d_in[34];
  const float* lin1b = (const float*)d_in[35];
  const float* lin2W = (const float*)d_in[36];
  const float* lin2b = (const float*)d_in[37];

  char* ws = (char*)d_ws;                        // 256B-aligned sections
  float* aggr = (float*)(ws + 0);                // N*128 f32
  float* hA   = (float*)(ws + 20480000);         // N*128 f32
  float* hB   = (float*)(ws + 40960000);         // N*128 f32
  float* p    = (float*)(ws + 61440000);         // G*384 f32
  float* q    = (float*)(ws + 61833216);         // G*384 f32
  float* b1f  = (float*)(ws + 62226432);         // 3*128 f32
  unsigned short* bf = (unsigned short*)(ws + 62227968);  // packed bf16 weights

  PackArgs pa;
  for (int l = 0; l < 3; ++l) {
    pa.W1[l] = cp(l, 2);  pa.b1[l]   = cp(l, 3);
    pa.gamma[l] = cp(l, 4); pa.beta[l] = cp(l, 5);
    pa.mean[l]  = cp(l, 6); pa.var[l]  = cp(l, 7);
    pa.W2[l] = cp(l, 8);
  }
  pa.lin1W = lin1W; pa.bf = bf; pa.b1f = b1f;
  pack_weights_kernel<<<930, 256, 0, stream>>>(pa);

  zero_kernel<<<(G_BATCH * 384 + 255) / 256, 256, 0, stream>>>(p, (size_t)G_BATCH * 384);

  const unsigned short* W1p[3] = { bf, bf + 8192, bf + 24576 };
  const unsigned short* W2p[3] = { bf + 40960, bf + 57344, bf + 73728 };
  const unsigned short* lin1p  = bf + 90112;

  const float* xin = x;
  float* houts[3] = { hA, hB, hA };
  for (int l = 0; l < 3; ++l) {
    int Cin = (l == 0) ? F_INCH : H_DIM;
    int lg  = (l == 0) ? 4 : 5;                         // log2(Cin/4)
    size_t na = (size_t)N_NODES * Cin;
    zero_kernel<<<(unsigned)((na + 255) / 256), 256, 0, stream>>>(aggr, na);
    long et = (long)N_EDGES << lg;
    edge_msg_kernel<<<(unsigned)((et + 255) / 256), 256, 0, stream>>>(
        xin, ei, ea, cp(l, 0), cp(l, 1), aggr, Cin, lg);
    node_mlp_kernel<<<N_NODES / 64, 256, 0, stream>>>(
        xin, aggr, W1p[l], b1f + 128 * l, W2p[l], cp(l, 9), houts[l], Cin);
    pool_kernel<<<(N_NODES * 32 + 255) / 256, 256, 0, stream>>>(
        houts[l], batch, p, 128 * l);
    xin = houts[l];
  }

  gemm_relu_kernel<<<dim3(G_BATCH / 64, 3), 256, 0, stream>>>(p, lin1p, lin1b, q, 384, 384);
  lin2_kernel<<<G_BATCH, 32, 0, stream>>>(q, lin2W, lin2b, (float*)d_out);
}
